// Int8Linear_89593017795344
// MI455X (gfx1250) — matrix operand order repaired and checked by reading, not compiled, not run on hardware
//
#include <hip/hip_runtime.h>
#include <hip/hip_fp16.h>
#include <stdint.h>

// ---- problem constants (match reference) ----
#define K_IN   4096
#define N_OUT  11008
#define M_ROWS 256        // B*S = 8*32

// ---- tiling ----
#define MT 4              // 16-row M-tiles per wave  -> 64 rows
#define NT 2              // 16-col N-tiles per wave  -> 32 cols
#define WAVES_N 8         // block covers 64 M x 256 N
#define BLOCK_THREADS 256 // 8 wave32

// ---- LDS staging of the shared A (x) slice ----
#define KC        256                    // K-chunk in halves (8 WMMA K-steps)
#define NCHUNK    (K_IN / KC)            // 16
#define ROW_H     (KC + 8)               // 264 halves = 528 B padded row stride
#define CHUNK_H   (64 * ROW_H)           // halves per buffer
#define CHUNK_B   (CHUNK_H * 2)          // bytes per buffer (33792)

typedef _Float16 v16h __attribute__((ext_vector_type(16)));
typedef _Float16 v8h  __attribute__((ext_vector_type(8)));
typedef _Float16 h2   __attribute__((ext_vector_type(2)));
typedef float    v8f  __attribute__((ext_vector_type(8)));
typedef uint32_t v4u  __attribute__((ext_vector_type(4)));

// Exact int8 -> f16 dequant, 5 VALU per 4 values:
//   u = b ^ 0x80 (excess-128); f16 bits 0x64|u == 1024 + 128 + b; minus 1152.0h -> b
__device__ __forceinline__ v16h dequant16(const int8_t* __restrict__ p) {
    const v4u d = *(const v4u*)p;                 // one global_load_b128 (16 int8)
    const h2 mk = { (_Float16)(-1152.0f), (_Float16)(-1152.0f) };
    v16h out;
#pragma unroll
    for (int j = 0; j < 4; ++j) {
        const uint32_t u  = d[j] ^ 0x80808080u;
        const uint32_t lo = __builtin_amdgcn_perm(0x64646464u, u, 0x05010400u);
        const uint32_t hi = __builtin_amdgcn_perm(0x64646464u, u, 0x05030402u);
        const h2 f01 = __builtin_bit_cast(h2, lo) + mk;
        const h2 f23 = __builtin_bit_cast(h2, hi) + mk;
        out[4 * j + 0] = f01[0];
        out[4 * j + 1] = f01[1];
        out[4 * j + 2] = f23[0];
        out[4 * j + 3] = f23[1];
    }
    return out;
}

// Async-DMA one 64-row x KC-half chunk of x into an LDS buffer (padded rows).
// 256 threads x 8 iters x 16 B; each wave covers exactly one 512 B row per iter.
__device__ __forceinline__ void stage_chunk(const _Float16* __restrict__ x,
                                            int m0, int kc0,
                                            uint32_t ldsBufByte, int tid) {
#pragma unroll
    for (int i = 0; i < 8; ++i) {
        const int s    = i * BLOCK_THREADS + tid;  // 16-byte segment index
        const int row  = s >> 5;                   // 32 segments per row
        const int kseg = s & 31;
        const uint32_t lds = ldsBufByte + (uint32_t)(row * (ROW_H * 2) + kseg * 16);
        const uint64_t g   = (uint64_t)(uintptr_t)(x + (size_t)(m0 + row) * K_IN
                                                     + kc0 + kseg * 8);
        asm volatile("global_load_async_to_lds_b128 %0, %1, off"
                     :: "v"(lds), "v"(g) : "memory");
    }
}

__global__ __launch_bounds__(BLOCK_THREADS)
void int8_linear_wmma_f16(const _Float16* __restrict__ x,     // [M_ROWS, K_IN]
                          const int8_t*   __restrict__ w,     // [N_OUT, K_IN]
                          const float*    __restrict__ scale, // [N_OUT]
                          const _Float16* __restrict__ bias,  // [N_OUT]
                          _Float16*       __restrict__ out)   // [M_ROWS, N_OUT]
{
    __shared__ _Float16 smemA[2 * CHUNK_H] __attribute__((aligned(16)));

    const int tid  = threadIdx.x;
    const int lane = tid & 31;
    const int wave = tid >> 5;
    const int l15  = lane & 15;   // row/col within a 16x16 tile
    const int h    = lane >> 4;   // lane half (K sub-range select)

    const int m0 = blockIdx.y * (MT * 16);                      // 64 rows shared by block
    const int n0 = blockIdx.x * (WAVES_N * NT * 16) + wave * (NT * 16);

    const uint32_t ldsBase = (uint32_t)(uintptr_t)&smemA[0];    // wave-relative LDS byte addr

    // B: lane covers col n0 + nt*16 + l15; K = k0 + h*16 + {0..15} (16 int8)
    const int8_t* wrow[NT];
#pragma unroll
    for (int nt = 0; nt < NT; ++nt)
        wrow[nt] = w + (size_t)(n0 + nt * 16 + l15) * K_IN;

    v8f acc[MT][NT];
#pragma unroll
    for (int mt = 0; mt < MT; ++mt)
#pragma unroll
        for (int nt = 0; nt < NT; ++nt)
            acc[mt][nt] = (v8f){};

    // prologue: DMA chunk 0 into buffer 0
    stage_chunk(x, m0, 0, ldsBase, tid);

    for (int c = 0; c < NCHUNK; ++c) {
        if (c + 1 < NCHUNK) {
            // DMA next chunk into the other buffer; overlaps this chunk's compute
            stage_chunk(x, m0, (c + 1) * KC, ldsBase + ((c + 1) & 1) * CHUNK_B, tid);
            asm volatile("s_wait_asynccnt 0x8" ::: "memory");   // chunk c resident (in-order)
        } else {
            asm volatile("s_wait_asynccnt 0x0" ::: "memory");
        }
        __syncthreads();                                        // chunk c visible to all waves

        const _Float16* sA = smemA + (c & 1) * CHUNK_H;

#pragma unroll
        for (int kk = 0; kk < KC / 32; ++kk) {
            // ---- A fragments from LDS (two ds_load_b128 each, conflict-free) ----
            v16h a[MT];
#pragma unroll
            for (int mt = 0; mt < MT; ++mt) {
                const int rbase = (mt * 16 + l15) * ROW_H + kk * 32;
                v8h lo = *(const v8h*)(sA + rbase + h * 16);       // K = h*16  + 0..7
                v8h hi = *(const v8h*)(sA + rbase + h * 16 + 8);   // K = h*16+8+ 0..7
#pragma unroll
                for (int i = 0; i < 8; ++i) { a[mt][i] = lo[i]; a[mt][8 + i] = hi[i]; }
            }

            const int k0  = c * KC + kk * 32;
            const int kpf = (k0 + 2048) & (K_IN - 1);              // branch-free stream hint

            // ---- B fragments: b128 int8 load + exact dequant, 4 WMMAs each ----
#pragma unroll
            for (int nt = 0; nt < NT; ++nt) {
                __builtin_prefetch(wrow[nt] + kpf, 0, 3);
                const v16h b = dequant16(wrow[nt] + k0 + h * 16);
#pragma unroll
                for (int mt = 0; mt < MT; ++mt)
                    acc[mt][nt] = __builtin_amdgcn_wmma_f32_16x16x32_f16(
                        /*neg_a=*/false, a[mt], /*neg_b=*/false, b,
                        /*c_mod=*/(short)0, acc[mt][nt],
                        /*reuse_a=*/false, /*reuse_b=*/false);
            }
        }
        __syncthreads();   // all waves done reading buf(c&1) before it is re-DMA'd
    }

    // ---- epilogue: out = acc*scale[n] + bias[n] in fp32, cast to fp16 ----
    // C layout: VGPR r, lane half h -> row M = 8*h + r; col N = lane&15.
#pragma unroll
    for (int nt = 0; nt < NT; ++nt) {
        const int   n  = n0 + nt * 16 + l15;
        const float s  = scale[n];
        const float bi = (float)bias[n];
#pragma unroll
        for (int mt = 0; mt < MT; ++mt) {
#pragma unroll
            for (int r = 0; r < 8; ++r) {
                const int m = m0 + mt * 16 + h * 8 + r;
                out[(size_t)m * N_OUT + n] = (_Float16)(acc[mt][nt][r] * s + bi);
            }
        }
    }
}

extern "C" void kernel_launch(void* const* d_in, const int* in_sizes, int n_in,
                              void* d_out, int out_size, void* d_ws, size_t ws_size,
                              hipStream_t stream) {
    (void)in_sizes; (void)n_in; (void)out_size; (void)d_ws; (void)ws_size;
    const _Float16* x     = (const _Float16*)d_in[0];
    const int8_t*   w     = (const int8_t*)  d_in[1];
    const float*    scale = (const float*)   d_in[2];
    const _Float16* bias  = (const _Float16*)d_in[3];
    _Float16*       out   = (_Float16*)      d_out;

    dim3 grid(N_OUT / (WAVES_N * NT * 16),   // 11008 / 256 = 43
              M_ROWS / (MT * 16));           // 256 / 64    = 4
    dim3 block(BLOCK_THREADS);
    int8_linear_wmma_f16<<<grid, block, 0, stream>>>(x, w, scale, bias, out);
}
